// RoiSelect_49100066128656
// MI455X (gfx1250) — matrix operand order repaired and checked by reading, not compiled, not run on hardware
//
#include <hip/hip_runtime.h>
#include <hip/hip_bf16.h>
#include <float.h>

typedef __attribute__((ext_vector_type(16))) _Float16 v16h;
typedef __attribute__((ext_vector_type(8)))  float    v8f;

#define BATCH   32
#define DIM     768
#define NH      12
#define HD      64
#define HH      56
#define WW      56
#define NTOK    (HH*WW + 1)   // 3137
#define NTILE   197           // ceil(3137/16)
#define NPAD    (NTILE*16)    // 3152
#define RS      7
#define HP      50
#define WP      50
#define PPOOL   (HP*WP)       // 2500
#define PRE_NMS 100
#define ROI_NUM 4
#define IOU_THR 0.3f
#define OUT1_ELEMS ((size_t)BATCH*197*DIM)

// ---------------------------------------------------------------------------
// Kernel A: w1[b,h,n] = <K[b,n,h,:], q0[b,h,:]>, w2[b,h,n] = <Q[b,n,h,:], k0[b,h,:]>
// via v_wmma_f32_16x16x32_f16. Each wave: one 16-token tile, D = tokens x heads.
// B operand is block-diagonal (head h occupies d in [h*64, h*64+64)), so each
// K=32 chunk has exactly one nonzero B column. 24 chunks x 2 = 48 WMMAs/wave.
// ---------------------------------------------------------------------------
__global__ __launch_bounds__(128)
void aw_wmma_kernel(const float* __restrict__ qkv,
                    float* __restrict__ w1, float* __restrict__ w2)
{
    const int b    = blockIdx.y;
    const int wave = threadIdx.x >> 5;
    const int lane = threadIdx.x & 31;
    const int tile = blockIdx.x * 4 + wave;
    if (tile >= NTILE) return;

    const int tb   = tile * 16;
    const int grp  = lane >> 4;      // lane-group (0: lanes 0-15, 1: lanes 16-31)
    const int mrow = lane & 15;      // A row (token-in-tile) == B column (head)
    const int tok  = tb + mrow;
    const bool tvalid = (tok < NTOK);

    const size_t stride = 3 * DIM;
    const float* base0 = qkv + (size_t)b * NTOK * stride;              // token 0
    const float* baseT = qkv + ((size_t)b * NTOK + (size_t)tok) * stride;

    v8f c1 = {};   // accum for w1  (A = K rows, B = q0 block-diag)
    v8f c2 = {};   // accum for w2  (A = Q rows, B = k0 block-diag)

    for (int chunk = 0; chunk < 24; ++chunk) {
        const int h     = chunk >> 1;
        const int cpart = chunk & 1;
        const int dbase = h * HD + cpart * 32;

        // B fragments: 16-bit B 32x16, lane-group 0: K 0-7 & 16-23, group 1: 8-15 & 24-31
        v16h bq = {}, bk = {};
        if (mrow == h) {
            const int koff = dbase + grp * 8;
            #pragma unroll
            for (int e = 0; e < 16; ++e) {
                const int kk = koff + e + (e >= 8 ? 8 : 0);
                bq[e] = (_Float16)base0[kk];          // q0 (first 768 chans)
                bk[e] = (_Float16)base0[DIM + kk];    // k0 (second 768)
            }
        }

        // A fragments: 16-bit A 16x32, lane-group 0: K 0-7 & 16-23, group 1: 8-15 & 24-31
        v16h aK = {}, aQ = {};
        if (tvalid) {
            const int koff = dbase + grp * 8;
            #pragma unroll
            for (int e = 0; e < 8; ++e) {
                aQ[e]     = (_Float16)baseT[koff + e];
                aQ[e + 8] = (_Float16)baseT[koff + 16 + e];
                aK[e]     = (_Float16)baseT[DIM + koff + e];
                aK[e + 8] = (_Float16)baseT[DIM + koff + 16 + e];
            }
        }

        c1 = __builtin_amdgcn_wmma_f32_16x16x32_f16(false, aK, false, bq,
                                                    (short)0, c1, false, false);
        c2 = __builtin_amdgcn_wmma_f32_16x16x32_f16(false, aQ, false, bk,
                                                    (short)0, c2, false, false);
    }

    // D layout: VGPR r, lane-group 0 -> token row r, group 1 -> row r+8; N = lane&15
    const int head = mrow;
    if (head < NH) {
        const size_t ob = (size_t)(b * NH + head) * NPAD;
        #pragma unroll
        for (int r = 0; r < 8; ++r) {
            const int t = tb + r + 8 * grp;   // < NPAD always
            w1[ob + t] = c1[r];
            w2[ob + t] = c2[r];
        }
    }
}

// ---------------------------------------------------------------------------
// Kernel B1: per (b,h) softmax stats (max + sum-exp) for w1 and w2 over NTOK.
// ---------------------------------------------------------------------------
__global__ __launch_bounds__(256)
void softmax_stats_kernel(const float* __restrict__ w1,
                          const float* __restrict__ w2,
                          float* __restrict__ stats)
{
    const int h = blockIdx.x, b = blockIdx.y;
    const size_t base = (size_t)(b * NH + h) * NPAD;
    __shared__ float r1[256], r2[256];
    const int tid = threadIdx.x;

    float m1 = -FLT_MAX, m2 = -FLT_MAX;
    for (int n = tid; n < NTOK; n += 256) {
        m1 = fmaxf(m1, w1[base + n]);
        m2 = fmaxf(m2, w2[base + n]);
    }
    r1[tid] = m1; r2[tid] = m2; __syncthreads();
    for (int s = 128; s > 0; s >>= 1) {
        if (tid < s) { r1[tid] = fmaxf(r1[tid], r1[tid+s]);
                       r2[tid] = fmaxf(r2[tid], r2[tid+s]); }
        __syncthreads();
    }
    m1 = r1[0]; m2 = r2[0]; __syncthreads();

    float s1 = 0.f, s2 = 0.f;
    for (int n = tid; n < NTOK; n += 256) {
        s1 += expf(w1[base + n] - m1);
        s2 += expf(w2[base + n] - m2);
    }
    r1[tid] = s1; r2[tid] = s2; __syncthreads();
    for (int s = 128; s > 0; s >>= 1) {
        if (tid < s) { r1[tid] += r1[tid+s]; r2[tid] += r2[tid+s]; }
        __syncthreads();
    }
    if (tid == 0) {
        float* st = stats + (size_t)(b * NH + h) * 4;
        st[0] = m1; st[1] = r1[0]; st[2] = m2; st[3] = r2[0];
    }
}

// ---------------------------------------------------------------------------
// Kernel B2: aw[b, n-1] = mean_h softmax(w1)*softmax(w2), tokens n=1..3136
// ---------------------------------------------------------------------------
__global__ __launch_bounds__(256)
void aweight_kernel(const float* __restrict__ w1, const float* __restrict__ w2,
                    const float* __restrict__ stats, float* __restrict__ aw)
{
    const int b = blockIdx.y;
    const int j = blockIdx.x * 256 + threadIdx.x;
    if (j >= HH * WW) return;
    const int n = j + 1;
    float acc = 0.f;
    #pragma unroll
    for (int h = 0; h < NH; ++h) {
        const size_t base = (size_t)(b * NH + h) * NPAD;
        const float* st = stats + (size_t)(b * NH + h) * 4;
        const float p1 = expf(w1[base + n] - st[0]) / st[1];
        const float p2 = expf(w2[base + n] - st[2]) / st[3];
        acc += p1 * p2;
    }
    aw[(size_t)b * (HH * WW) + j] = acc * (1.0f / NH);
}

// ---------------------------------------------------------------------------
// Kernel C1: 7x7 VALID mean pooling over (56,56) -> (50,50) per batch.
// ---------------------------------------------------------------------------
__global__ __launch_bounds__(256)
void roi_score_kernel(const float* __restrict__ aw, float* __restrict__ scores)
{
    const int b = blockIdx.x;
    __shared__ float tile[HH * WW];   // 12.25 KB
    for (int i = threadIdx.x; i < HH * WW; i += 256)
        tile[i] = aw[(size_t)b * HH * WW + i];
    __syncthreads();
    for (int p = threadIdx.x; p < PPOOL; p += 256) {
        const int i = p / WP, j = p % WP;
        float s = 0.f;
        for (int dy = 0; dy < RS; ++dy)
            for (int dx = 0; dx < RS; ++dx)
                s += tile[(i + dy) * WW + (j + dx)];
        scores[(size_t)b * PPOOL + p] = s * (1.0f / (RS * RS));
    }
}

// ---------------------------------------------------------------------------
// Kernel C2: per-batch top-100 (iterative argmax), greedy NMS, select 4 rois.
// ---------------------------------------------------------------------------
__global__ __launch_bounds__(256)
void nms_kernel(const float* __restrict__ scores_in, float* __restrict__ rois_out)
{
    const int b = blockIdx.x;
    const int tid = threadIdx.x;
    __shared__ float sc[PPOOL];
    __shared__ float rv[256];
    __shared__ int   ri[256];
    __shared__ int   topIdx[PRE_NMS];
    __shared__ int   keep[PRE_NMS];

    for (int i = tid; i < PPOOL; i += 256) sc[i] = scores_in[(size_t)b * PPOOL + i];
    __syncthreads();

    // top-100, descending, ties -> lowest index (matches top_k)
    for (int k = 0; k < PRE_NMS; ++k) {
        float bv = -FLT_MAX; int bi = PPOOL;
        for (int i = tid; i < PPOOL; i += 256) {
            const float v = sc[i];
            if (v > bv || (v == bv && i < bi)) { bv = v; bi = i; }
        }
        rv[tid] = bv; ri[tid] = bi;
        __syncthreads();
        for (int s = 128; s > 0; s >>= 1) {
            if (tid < s) {
                const float ov = rv[tid+s]; const int oi = ri[tid+s];
                if (ov > rv[tid] || (ov == rv[tid] && oi < ri[tid])) { rv[tid]=ov; ri[tid]=oi; }
            }
            __syncthreads();
        }
        if (tid == 0) { topIdx[k] = ri[0]; sc[ri[0]] = -FLT_MAX; }
        __syncthreads();
    }

    if (tid < PRE_NMS) keep[tid] = 1;
    __syncthreads();

    // greedy NMS (boxes: x1=idx/50, y1=idx%50, x2=x1+7, y2=y1+7; area=49)
    for (int i = 0; i < PRE_NMS; ++i) {
        if (keep[i]) {
            const float ax1 = (float)(topIdx[i] / WP);
            const float ay1 = (float)(topIdx[i] % WP);
            const float ax2 = ax1 + RS, ay2 = ay1 + RS;
            for (int j = tid; j < PRE_NMS; j += 256) {
                if (j > i && keep[j]) {
                    const float bx1 = (float)(topIdx[j] / WP);
                    const float by1 = (float)(topIdx[j] % WP);
                    const float bx2 = bx1 + RS, by2 = by1 + RS;
                    const float iw = fmaxf(0.f, fminf(ax2, bx2) - fmaxf(ax1, bx1));
                    const float ih = fmaxf(0.f, fminf(ay2, by2) - fmaxf(ay1, by1));
                    const float inter = iw * ih;
                    const float iou = inter / (98.0f - inter);  // 49 + 49 - inter
                    if (iou > IOU_THR) keep[j] = 0;
                }
            }
        }
        __syncthreads();
    }

    if (tid == 0) {
        int cnt = 0;
        for (int i = 0; i < PRE_NMS; ++i) cnt += keep[i];
        int sel[ROI_NUM]; int w = 0;
        for (int i = 0; i < PRE_NMS && w < ROI_NUM; ++i) if (keep[i])  sel[w++] = i;
        for (int i = 0; i < PRE_NMS && w < ROI_NUM; ++i) if (!keep[i]) sel[w++] = i;
        for (int r = 0; r < ROI_NUM; ++r) {
            float x1 = 0.f, y1 = 0.f, x2 = 0.f, y2 = 0.f;
            if (r < cnt) {
                const int idx = topIdx[sel[r]];
                x1 = (float)(idx / WP); y1 = (float)(idx % WP);
                x2 = x1 + RS;           y2 = y1 + RS;
            }
            float* ro = rois_out + (size_t)(b * ROI_NUM + r) * 5;
            ro[0] = (float)b; ro[1] = x1; ro[2] = y1; ro[3] = x2; ro[4] = y2;
        }
    }
}

// ---------------------------------------------------------------------------
// Kernel D: ROI max-pool, writing directly into transposed output layout:
// out[b, 1 + r*49 + ph*7+pw, ch] = max over masked 8x8 patch.
// Replicates the reference's integer bin logic and -inf padding at index 56.
// ---------------------------------------------------------------------------
__global__ __launch_bounds__(256)
void roi_pool_kernel(const float* __restrict__ x, const float* __restrict__ rois,
                     float* __restrict__ out)
{
    const int roi = blockIdx.x;                 // 0..127
    const int b = roi / ROI_NUM, r = roi % ROI_NUM;
    const float* rp = rois + (size_t)roi * 5;
    const int x1 = (int)roundf(rp[1]);
    const int y1 = (int)roundf(rp[2]);
    const int x2 = (int)roundf(rp[3]);
    const int y2 = (int)roundf(rp[4]);
    const int rw = max(x2 - x1 + 1, 1);
    const int rh = max(y2 - y1 + 1, 1);

    int hs[RS], he[RS], wsb[RS], web[RS];
    #pragma unroll
    for (int pb = 0; pb < RS; ++pb) {
        hs[pb]  = pb * rh / RS;
        he[pb]  = ((pb + 1) * rh + RS - 1) / RS;
        wsb[pb] = pb * rw / RS;
        web[pb] = ((pb + 1) * rw + RS - 1) / RS;
    }

    for (int ch = threadIdx.x; ch < DIM; ch += 256) {
        float patch[RS + 1][RS + 1];
        // reference quirk: rows indexed by roi[2] (y1), cols by roi[1] (x1)
        #pragma unroll
        for (int ri2 = 0; ri2 <= RS; ++ri2) {
            const int row = min(y1 + ri2, HH);
            #pragma unroll
            for (int ci = 0; ci <= RS; ++ci) {
                const int col = min(x1 + ci, WW);
                float v = -FLT_MAX;
                if (row < HH && col < WW)
                    v = x[((size_t)b * NTOK + 1 + (size_t)row * WW + col) * DIM + ch];
                patch[ri2][ci] = v;
            }
        }
        for (int ph = 0; ph < RS; ++ph) {
            for (int pw = 0; pw < RS; ++pw) {
                float m = -FLT_MAX;
                for (int ri2 = hs[ph]; ri2 < he[ph] && ri2 <= RS; ++ri2)
                    for (int ci = wsb[pw]; ci < web[pw] && ci <= RS; ++ci)
                        m = fmaxf(m, patch[ri2][ci]);
                out[((size_t)b * 197 + 1 + r * (RS * RS) + ph * RS + pw) * DIM + ch] = m;
            }
        }
    }
}

// ---------------------------------------------------------------------------
// Kernel E: copy cls token  out[b, 0, :] = x[b, 0, :]
// ---------------------------------------------------------------------------
__global__ __launch_bounds__(256)
void cls_copy_kernel(const float* __restrict__ x, float* __restrict__ out)
{
    const int i = blockIdx.x * 256 + threadIdx.x;
    if (i >= BATCH * DIM) return;
    const int b = i / DIM, ch = i % DIM;
    out[(size_t)b * 197 * DIM + ch] = x[(size_t)b * NTOK * DIM + ch];
}

// ---------------------------------------------------------------------------
extern "C" void kernel_launch(void* const* d_in, const int* in_sizes, int n_in,
                              void* d_out, int out_size, void* d_ws, size_t ws_size,
                              hipStream_t stream)
{
    const float* x   = (const float*)d_in[0];   // (B, 3137, 768)
    const float* qkv = (const float*)d_in[1];   // (B, 3137, 2304)
    float* out = (float*)d_out;
    float* wsp = (float*)d_ws;

    // workspace layout (floats): ~10.4 MB total
    float* w1 = wsp;                                   // B*NH*NPAD
    float* w2 = w1 + (size_t)BATCH * NH * NPAD;        // B*NH*NPAD
    float* st = w2 + (size_t)BATCH * NH * NPAD;        // B*NH*4
    float* aw = st + (size_t)BATCH * NH * 4;           // B*3136
    float* sc = aw + (size_t)BATCH * HH * WW;          // B*2500
    float* rois = out + OUT1_ELEMS;                    // second tuple output (128*5)

    aw_wmma_kernel<<<dim3((NTILE + 3) / 4, BATCH), 128, 0, stream>>>(qkv, w1, w2);
    softmax_stats_kernel<<<dim3(NH, BATCH), 256, 0, stream>>>(w1, w2, st);
    aweight_kernel<<<dim3((HH * WW + 255) / 256, BATCH), 256, 0, stream>>>(w1, w2, st, aw);
    roi_score_kernel<<<BATCH, 256, 0, stream>>>(aw, sc);
    nms_kernel<<<BATCH, 256, 0, stream>>>(sc, rois);
    roi_pool_kernel<<<BATCH * ROI_NUM, 256, 0, stream>>>(x, rois, out);
    cls_copy_kernel<<<(BATCH * DIM + 255) / 256, 256, 0, stream>>>(x, out);
}
